// TARDISCell_3770981286447
// MI455X (gfx1250) — compile-verified, matrix operand order checked by reading, not run on hardware
//
#include <hip/hip_runtime.h>
#include <hip/hip_bf16.h>
#include <math.h>

typedef __attribute__((ext_vector_type(16))) __bf16 v16bf;
typedef __attribute__((ext_vector_type(8)))  float  v8f;

#define LDST 48  // LDS row stride in bf16 elements (96B: 32B-aligned frag reads, bank-spread)

// Permute K index so each lane's 16 fragment elements are contiguous in LDS.
// Lane half h reads columns [h*16, h*16+16): cols 0-7 <- k 0-7, 8-15 <- k 16-23,
// 16-23 <- k 8-15, 24-31 <- k 24-31  (ISA 16-bit A/B 16x32 layout).
__device__ __forceinline__ int kperm(int kk) {
  return (kk & 7) | (((kk >> 3) & 1) << 4) | ((kk >> 4) << 3);
}
__device__ __forceinline__ float sigm(float v) { return 1.0f / (1.0f + __expf(-v)); }

// ---------------------------------------------------------------------------
// Kernel 1: i/f/cand gates GEMM (bf16 WMMA) + c_new epilogue.
// Block tile 128x128, 8 waves (4 row-groups x 2 col-groups), wave tile 32x64.
// A = [x | h | c] (K=3072); cand only uses K<2048.
// ---------------------------------------------------------------------------
__global__ __launch_bounds__(256) void tardis_gates_kernel(
    const float* __restrict__ x, const float* __restrict__ h, const float* __restrict__ c,
    const float* __restrict__ Wi, const float* __restrict__ bias_i,
    const float* __restrict__ Wf, const float* __restrict__ bias_f,
    const float* __restrict__ Wc, const float* __restrict__ bias_c,
    float* __restrict__ out)
{
  __shared__ __bf16 As[128 * LDST];
  __shared__ __bf16 Bi[128 * LDST];
  __shared__ __bf16 Bf[128 * LDST];
  __shared__ __bf16 Bc[128 * LDST];

  const int rowBase = blockIdx.y * 128;
  const int colBase = blockIdx.x * 128;
  const int tid  = threadIdx.x;
  const int wid  = tid >> 5, lane = tid & 31;
  const int wrow = wid >> 1, wcol = wid & 1;
  const int lm   = lane & 15, half = lane >> 4;

  v8f acc_i[2][4], acc_f[2][4], acc_c[2][4];
  const v8f vzero = {0.f, 0.f, 0.f, 0.f, 0.f, 0.f, 0.f, 0.f};
#pragma unroll
  for (int mt = 0; mt < 2; ++mt)
#pragma unroll
    for (int nt = 0; nt < 4; ++nt) {
      acc_i[mt][nt] = vzero; acc_f[mt][nt] = vzero; acc_c[mt][nt] = vzero;
    }

  for (int kb = 0; kb < 3072; kb += 32) {
    __syncthreads();
    // ---- stage A tile (128 rows x 32 k), f32 -> bf16, K-permuted ----
#pragma unroll
    for (int i = 0; i < 16; ++i) {
      int idx = tid + i * 256;          // 0..4095
      int m   = idx >> 5;
      int kk  = idx & 31;
      int k   = kb + kk;
      int row = rowBase + m;
      float v;
      if (k < 1024)      v = x[row * 1024 + k];
      else if (k < 2048) v = h[row * 1024 + (k - 1024)];
      else               v = c[row * 1024 + (k - 2048)];
      As[m * LDST + kperm(kk)] = (__bf16)v;
    }
    // ---- stage B tiles (32 k x 128 n), stored N-major K-permuted ----
#pragma unroll
    for (int i = 0; i < 16; ++i) {
      int idx = tid + i * 256;
      int kk  = idx >> 7;               // 0..31
      int nn  = idx & 127;
      int k   = kb + kk;
      int col = colBase + nn;
      Bi[nn * LDST + kperm(kk)] = (__bf16)Wi[k * 1024 + col];
      Bf[nn * LDST + kperm(kk)] = (__bf16)Wf[k * 1024 + col];
      if (kb < 2048) Bc[nn * LDST + kperm(kk)] = (__bf16)Wc[k * 1024 + col];
    }
    __syncthreads();

    // ---- fragments + WMMA ----
    v16bf afrag[2];
#pragma unroll
    for (int mt = 0; mt < 2; ++mt)
      afrag[mt] = *(const v16bf*)&As[(wrow * 32 + mt * 16 + lm) * LDST + half * 16];

    v16bf bfrag[4];
#pragma unroll
    for (int nt = 0; nt < 4; ++nt)
      bfrag[nt] = *(const v16bf*)&Bi[(wcol * 64 + nt * 16 + lm) * LDST + half * 16];
#pragma unroll
    for (int mt = 0; mt < 2; ++mt)
#pragma unroll
      for (int nt = 0; nt < 4; ++nt)
        acc_i[mt][nt] = __builtin_amdgcn_wmma_f32_16x16x32_bf16(
            false, afrag[mt], false, bfrag[nt], (short)0, acc_i[mt][nt], false, false);

#pragma unroll
    for (int nt = 0; nt < 4; ++nt)
      bfrag[nt] = *(const v16bf*)&Bf[(wcol * 64 + nt * 16 + lm) * LDST + half * 16];
#pragma unroll
    for (int mt = 0; mt < 2; ++mt)
#pragma unroll
      for (int nt = 0; nt < 4; ++nt)
        acc_f[mt][nt] = __builtin_amdgcn_wmma_f32_16x16x32_bf16(
            false, afrag[mt], false, bfrag[nt], (short)0, acc_f[mt][nt], false, false);

    if (kb < 2048) {
#pragma unroll
      for (int nt = 0; nt < 4; ++nt)
        bfrag[nt] = *(const v16bf*)&Bc[(wcol * 64 + nt * 16 + lm) * LDST + half * 16];
#pragma unroll
      for (int mt = 0; mt < 2; ++mt)
#pragma unroll
        for (int nt = 0; nt < 4; ++nt)
          acc_c[mt][nt] = __builtin_amdgcn_wmma_f32_16x16x32_bf16(
              false, afrag[mt], false, bfrag[nt], (short)0, acc_c[mt][nt], false, false);
    }
  }

  // ---- epilogue: c_new = sigmoid(f)*c + sigmoid(i)*tanh(cand) ----
#pragma unroll
  for (int mt = 0; mt < 2; ++mt)
#pragma unroll
    for (int nt = 0; nt < 4; ++nt) {
      int col  = colBase + wcol * 64 + nt * 16 + lm;
      float bi = bias_i[col], bff = bias_f[col], bcc = bias_c[col];
#pragma unroll
      for (int v = 0; v < 8; ++v) {
        int row  = rowBase + wrow * 32 + mt * 16 + v + 8 * half;
        float iv = sigm(acc_i[mt][nt][v] + bi);
        float fv = sigm(acc_f[mt][nt][v] + bff);
        float cp = tanhf(acc_c[mt][nt][v] + bcc);
        float cn = fv * c[row * 1024 + col] + iv * cp;
        out[row * 2050 + 1024 + col] = cn;
      }
    }
}

// ---------------------------------------------------------------------------
// Kernel 2: o-gate GEMM over [x | h | c_new] (c_new streamed from d_out),
// epilogue h_new = sigmoid(o)*tanh(c_new).
// ---------------------------------------------------------------------------
__global__ __launch_bounds__(256) void tardis_ogate_kernel(
    const float* __restrict__ x, const float* __restrict__ h,
    const float* __restrict__ Wo, const float* __restrict__ bias_o,
    float* __restrict__ out)
{
  __shared__ __bf16 As[128 * LDST];
  __shared__ __bf16 Bs[128 * LDST];

  const int rowBase = blockIdx.y * 128;
  const int colBase = blockIdx.x * 128;
  const int tid  = threadIdx.x;
  const int wid  = tid >> 5, lane = tid & 31;
  const int wrow = wid >> 1, wcol = wid & 1;
  const int lm   = lane & 15, half = lane >> 4;

  v8f acc[2][4];
  const v8f vzero = {0.f, 0.f, 0.f, 0.f, 0.f, 0.f, 0.f, 0.f};
#pragma unroll
  for (int mt = 0; mt < 2; ++mt)
#pragma unroll
    for (int nt = 0; nt < 4; ++nt) acc[mt][nt] = vzero;

  for (int kb = 0; kb < 3072; kb += 32) {
    __syncthreads();
#pragma unroll
    for (int i = 0; i < 16; ++i) {
      int idx = tid + i * 256;
      int m   = idx >> 5;
      int kk  = idx & 31;
      int k   = kb + kk;
      int row = rowBase + m;
      float v;
      if (k < 1024)      v = x[row * 1024 + k];
      else if (k < 2048) v = h[row * 1024 + (k - 1024)];
      else               v = out[row * 2050 + 1024 + (k - 2048)];  // c_new
      As[m * LDST + kperm(kk)] = (__bf16)v;
    }
#pragma unroll
    for (int i = 0; i < 16; ++i) {
      int idx = tid + i * 256;
      int kk  = idx >> 7;
      int nn  = idx & 127;
      Bs[nn * LDST + kperm(kk)] = (__bf16)Wo[(kb + kk) * 1024 + colBase + nn];
    }
    __syncthreads();

    v16bf afrag[2];
#pragma unroll
    for (int mt = 0; mt < 2; ++mt)
      afrag[mt] = *(const v16bf*)&As[(wrow * 32 + mt * 16 + lm) * LDST + half * 16];
    v16bf bfrag[4];
#pragma unroll
    for (int nt = 0; nt < 4; ++nt)
      bfrag[nt] = *(const v16bf*)&Bs[(wcol * 64 + nt * 16 + lm) * LDST + half * 16];
#pragma unroll
    for (int mt = 0; mt < 2; ++mt)
#pragma unroll
      for (int nt = 0; nt < 4; ++nt)
        acc[mt][nt] = __builtin_amdgcn_wmma_f32_16x16x32_bf16(
            false, afrag[mt], false, bfrag[nt], (short)0, acc[mt][nt], false, false);
  }

#pragma unroll
  for (int mt = 0; mt < 2; ++mt)
#pragma unroll
    for (int nt = 0; nt < 4; ++nt) {
      int col  = colBase + wcol * 64 + nt * 16 + lm;
      float bo = bias_o[col];
#pragma unroll
      for (int v = 0; v < 8; ++v) {
        int row  = rowBase + wrow * 32 + mt * 16 + v + 8 * half;
        float ov = sigm(acc[mt][nt][v] + bo);
        float cn = out[row * 2050 + 1024 + col];
        out[row * 2050 + col] = ov * tanhf(cn);
      }
    }
}

// ---------------------------------------------------------------------------
// Kernel 3: per-row attention: micro GEMV, Gumbel-softmax (M=16), sampled,
// alpha/beta gumbel-sigmoid. One 256-thread block per batch row.
// ---------------------------------------------------------------------------
__global__ __launch_bounds__(256) void tardis_attn_kernel(
    const float* __restrict__ x, const float* __restrict__ memory,
    const float* __restrict__ u_gs, const float* __restrict__ u_a, const float* __restrict__ u_b,
    const float* __restrict__ Wm, const float* __restrict__ bm,
    const float* __restrict__ wah, const float* __restrict__ wax, const float* __restrict__ war,
    const float* __restrict__ wbh, const float* __restrict__ wbx, const float* __restrict__ wbr,
    float* __restrict__ out)
{
  const int r = blockIdx.x;
  const int t = threadIdx.x;
  __shared__ float hn[1024];
  __shared__ float red[256];
  __shared__ float redb[256];
  __shared__ float mic[64];
  __shared__ float attn[16];
  __shared__ float smp[64];

  float* orow = out + (size_t)r * 2050;
#pragma unroll
  for (int i = 0; i < 4; ++i) hn[t + i * 256] = orow[t + i * 256];
  __syncthreads();

  // micro = h_new @ W_m + b_m   (4 threads per output column)
  {
    int j = t & 63, part = t >> 6;
    float s = 0.f;
    for (int k = part * 256; k < part * 256 + 256; ++k) s += hn[k] * Wm[k * 64 + j];
    red[t] = s;
  }
  __syncthreads();
  if (t < 64) mic[t] = red[t] + red[64 + t] + red[128 + t] + red[192 + t] + bm[t];
  __syncthreads();

  // logits[m] = memory[r,m,:] . micro  (16 threads per slot)
  {
    int m = t >> 4, kq = (t & 15) * 4;
    const float* mrow = memory + (size_t)r * 1024 + m * 64;
    float s = 0.f;
#pragma unroll
    for (int q = 0; q < 4; ++q) s += mrow[kq + q] * mic[kq + q];
    red[t] = s;
  }
  __syncthreads();
  if (t < 16) {
    float s = 0.f;
#pragma unroll
    for (int q = 0; q < 16; ++q) s += red[t * 16 + q];
    float u = u_gs[r * 16 + t];
    float g = -__logf(-__logf(u));          // Gumbel(0,1)
    attn[t] = s + g;                        // TAU = 1
  }
  __syncthreads();
  if (t == 0) {                             // softmax over 16 slots
    float mx = attn[0];
    for (int m = 1; m < 16; ++m) mx = fmaxf(mx, attn[m]);
    float sum = 0.f;
    for (int m = 0; m < 16; ++m) { float e = __expf(attn[m] - mx); attn[m] = e; sum += e; }
    float inv = 1.f / sum;
    for (int m = 0; m < 16; ++m) attn[m] *= inv;
  }
  __syncthreads();
  if (t < 64) {                             // sampled = attn @ memory + 1
    float s = 1.0f;
    const float* mrow = memory + (size_t)r * 1024;
#pragma unroll
    for (int m = 0; m < 16; ++m) s += mrow[m * 64 + t] * attn[m];
    smp[t] = s;
  }
  __syncthreads();

  // alpha / beta reductions
  {
    const float* xr = x + (size_t)r * 1024;
    float pa = 0.f, pb = 0.f;
    for (int k = t; k < 1024; k += 256) {
      float hv = hn[k], xv = xr[k];
      pa += hv * wah[k] + xv * wax[k];
      pb += hv * wbh[k] + xv * wbx[k];
    }
    if (t < 64) { pa += smp[t] * war[t]; pb += smp[t] * wbr[t]; }
    red[t] = pa; redb[t] = pb;
  }
  __syncthreads();
  for (int s = 128; s > 0; s >>= 1) {
    if (t < s) { red[t] += red[t + s]; redb[t] += redb[t + s]; }
    __syncthreads();
  }
  if (t == 0) {
    float ua = u_a[r], ub = u_b[r];
    float la = __logf(ua) - log1pf(-ua);
    float lb = __logf(ub) - log1pf(-ub);
    orow[2048] = sigm(red[0] + la);
    orow[2049] = sigm(redb[0] + lb);
  }
}

extern "C" void kernel_launch(void* const* d_in, const int* in_sizes, int n_in,
                              void* d_out, int out_size, void* d_ws, size_t ws_size,
                              hipStream_t stream) {
  (void)in_sizes; (void)n_in; (void)out_size; (void)d_ws; (void)ws_size;
  const float* x      = (const float*)d_in[0];
  const float* h      = (const float*)d_in[1];
  const float* c      = (const float*)d_in[2];
  const float* memory = (const float*)d_in[3];
  const float* u_gs   = (const float*)d_in[4];
  const float* u_a    = (const float*)d_in[5];
  const float* u_b    = (const float*)d_in[6];
  const float* W_i    = (const float*)d_in[7];
  const float* b_i    = (const float*)d_in[8];
  const float* W_f    = (const float*)d_in[9];
  const float* b_f    = (const float*)d_in[10];
  const float* W_o    = (const float*)d_in[11];
  const float* b_o    = (const float*)d_in[12];
  const float* W_cand = (const float*)d_in[13];
  const float* b_cand = (const float*)d_in[14];
  const float* W_m    = (const float*)d_in[15];
  const float* b_m    = (const float*)d_in[16];
  const float* w_a_h  = (const float*)d_in[17];
  const float* w_a_x  = (const float*)d_in[18];
  const float* w_a_r  = (const float*)d_in[19];
  const float* w_b_h  = (const float*)d_in[20];
  const float* w_b_x  = (const float*)d_in[21];
  const float* w_b_r  = (const float*)d_in[22];
  float* out = (float*)d_out;

  dim3 grid(1024 / 128, 8192 / 128);  // (N-blocks=8, M-blocks=64)
  dim3 blk(256);
  tardis_gates_kernel<<<grid, blk, 0, stream>>>(x, h, c, W_i, b_i, W_f, b_f,
                                                W_cand, b_cand, out);
  tardis_ogate_kernel<<<grid, blk, 0, stream>>>(x, h, W_o, b_o, out);
  tardis_attn_kernel<<<8192, 256, 0, stream>>>(x, memory, u_gs, u_a, u_b,
                                               W_m, b_m, w_a_h, w_a_x, w_a_r,
                                               w_b_h, w_b_x, w_b_r, out);
}
